// STRNNCell_64218351010369
// MI455X (gfx1250) — matrix operand order repaired and checked by reading, not compiled, not run on hardware
//
#include <hip/hip_runtime.h>

// STRNNCell on gfx1250 (MI455X): slot-bucketed WMMA GEMMs.
// B=256, C=32, I=128, J=64, H=128, slots=49.

#define Bn 256
#define Cn 32
#define In 128
#define Jn 64
#define Hn 128
#define NSLOT 49
#define NPAIR (Bn * Cn)   // 8192

typedef __attribute__((ext_vector_type(16))) _Float16 v16h;
typedef __attribute__((ext_vector_type(8)))  float    v8f;
typedef __attribute__((ext_vector_type(4)))  float    f32x4;

__device__ __forceinline__ v8f wmma16(v16h a, v16h b, v8f c) {
  // D(16x16 f32) = A(16x32 f16) * B(32x16 f16) + C
  return __builtin_amdgcn_wmma_f32_16x16x32_f16(false, a, false, b, (short)0, c,
                                                false, false);
}

// B-tile load: lane n=lane&15 holds column n; lanes 0-15 hold K=k0..k0+7 and
// k0+16..k0+23, lanes 16-31 hold K=k0+8..15 and k0+24..31 (same as A).
// With the weight stored TRANSPOSED (row n, contiguous K) this is two
// global_load_b128 at p and p+16.
__device__ __forceinline__ v16h load_b16(const _Float16* p) {
  v16h r;
  ((f32x4*)&r)[0] = ((const f32x4*)p)[0];
  ((f32x4*)&r)[1] = *(const f32x4*)(p + 16);
  return r;
}

// A-tile load: lane m=lane&15 holds row m. Lanes 0-15: K=k0..k0+7 (v0-3) and
// K=k0+16..k0+23 (v4-7); lanes 16-31: K=k0+8..15 and k0+24..31. So per lane:
// two b128 loads at (row + k0 + sub) and (+16), sub = (lane&16)?8:0.
__device__ __forceinline__ v16h load_a16(const _Float16* p) {
  v16h r;
  ((f32x4*)&r)[0] = *(const f32x4*)(p);
  ((f32x4*)&r)[1] = *(const f32x4*)(p + 16);
  return r;
}

// ---------------- prep kernels ----------------

__global__ void k_init(int* tcnt, int* dcnt) {
  int i = blockIdx.x * blockDim.x + threadIdx.x;
  if (i < NSLOT) { tcnt[i] = 0; dcnt[i] = 0; }
}

__global__ void k_cvt(const float* __restrict__ src, _Float16* __restrict__ dst, int n) {
  int i = blockIdx.x * blockDim.x + threadIdx.x;
  if (i < n) dst[i] = (_Float16)src[i];
}

// dst[(s*cols + c)*rows + r] = (f16) src[(s*rows + r)*cols + c]
__global__ void k_cvtT(const float* __restrict__ src, _Float16* __restrict__ dst,
                       int slots, int rows, int cols) {
  int i = blockIdx.x * blockDim.x + threadIdx.x;
  int tot = slots * rows * cols;
  if (i >= tot) return;
  int s = i / (rows * cols);
  int rem = i % (rows * cols);
  int r = rem / cols, c = rem % cols;
  dst[((size_t)s * cols + c) * rows + r] = (_Float16)src[i];
}

__global__ void k_count(const int* __restrict__ tc, const int* __restrict__ dc,
                        int* tcnt, int* dcnt) {
  int i = blockIdx.x * blockDim.x + threadIdx.x;
  if (i < NPAIR) {
    atomicAdd(&tcnt[tc[i]], 1);
    atomicAdd(&dcnt[dc[i]], 1);
  }
}

__global__ void k_scan(const int* __restrict__ tcnt, int* toff, int* tcur,
                       const int* __restrict__ dcnt, int* doff, int* dcur) {
  if (blockIdx.x == 0 && threadIdx.x == 0) {
    int a = 0, b = 0;
    for (int s = 0; s < NSLOT; ++s) {
      toff[s] = a; tcur[s] = a; a += tcnt[s];
      doff[s] = b; dcur[s] = b; b += dcnt[s];
    }
  }
}

__global__ void k_fill(const int* __restrict__ tc, const int* __restrict__ dc,
                       int* tcur, int* tlist, int* dcur, int* dlist) {
  int i = blockIdx.x * blockDim.x + threadIdx.x;
  if (i < NPAIR) {
    int p = atomicAdd(&tcur[tc[i]], 1); tlist[p] = i;
    int q = atomicAdd(&dcur[dc[i]], 1); dlist[q] = i;
  }
}

// ---------------- GEMM 1: xt[bc, 0:64] = x[bc, 0:128] @ TW[t_bc] ----------------
// grid (49, 16), block 32 (one wave). Rows of a slot's bucket tiled by 16.
__global__ __launch_bounds__(32)
void k_gemm1(const _Float16* __restrict__ x16, const _Float16* __restrict__ twT,
             const int* __restrict__ toff, const int* __restrict__ tcnt,
             const int* __restrict__ tlist, _Float16* __restrict__ xt16) {
  const int s = blockIdx.x;
  const int lane = threadIdx.x;
  const int start = toff[s];
  const int count = tcnt[s];
  const int nB   = lane & 15;
  const int subA = (lane & 16) ? 8 : 0;
  const int kbB  = (lane & 16) ? 8 : 0;
  const int rhi  = (lane & 16) ? 8 : 0;

  for (int t = blockIdx.y; t * 16 < count; t += gridDim.y) {
    const int rowBase = t * 16;
    int r = rowBase + (lane & 15);
    if (r >= count) r = count - 1;             // pad with a valid row; not stored
    const int bcA = tlist[start + r];
    const _Float16* arow = x16 + (size_t)bcA * In + subA;

    v8f acc0 = {}, acc1 = {}, acc2 = {}, acc3 = {};
    #pragma unroll
    for (int k0 = 0; k0 < In; k0 += 32) {
      v16h a = load_a16(arow + k0);
      const _Float16* bbase = twT + ((size_t)s * Jn + nB) * In + k0 + kbB;
      v16h b0 = load_b16(bbase);               // j-tile 0  (rows nB+0..)
      v16h b1 = load_b16(bbase + 16 * In);     // j-tile 16
      v16h b2 = load_b16(bbase + 32 * In);     // j-tile 32
      v16h b3 = load_b16(bbase + 48 * In);     // j-tile 48
      acc0 = wmma16(a, b0, acc0);
      acc1 = wmma16(a, b1, acc1);
      acc2 = wmma16(a, b2, acc2);
      acc3 = wmma16(a, b3, acc3);
    }
    #pragma unroll
    for (int v = 0; v < 8; ++v) {
      int rr = rowBase + v + rhi;              // D: VGPR v -> row v (+8 for hi lanes)
      if (rr < count) {
        int bc = tlist[start + rr];
        _Float16* op = xt16 + (size_t)bc * Jn + nB;
        op[0]  = (_Float16)acc0[v];
        op[16] = (_Float16)acc1[v];
        op[32] = (_Float16)acc2[v];
        op[48] = (_Float16)acc3[v];
      }
    }
  }
}

// ---------------- GEMM 2: xch[bc, 0:128] = mask * (xt[bc] @ DW[d_bc]) ----------------
__global__ __launch_bounds__(32)
void k_gemm2(const _Float16* __restrict__ xt16, const _Float16* __restrict__ dwT,
             const int* __restrict__ doff, const int* __restrict__ dcnt,
             const int* __restrict__ dlist, const unsigned char* __restrict__ mask,
             float* __restrict__ xch) {
  const int s = blockIdx.x;
  const int lane = threadIdx.x;
  const int start = doff[s];
  const int count = dcnt[s];
  const int nB   = lane & 15;
  const int subA = (lane & 16) ? 8 : 0;
  const int kbB  = (lane & 16) ? 8 : 0;
  const int rhi  = (lane & 16) ? 8 : 0;

  for (int t = blockIdx.y; t * 16 < count; t += gridDim.y) {
    const int rowBase = t * 16;
    int r = rowBase + (lane & 15);
    if (r >= count) r = count - 1;
    const int bcA = dlist[start + r];
    const _Float16* arow = xt16 + (size_t)bcA * Jn + subA;

    v8f acc[8];
    #pragma unroll
    for (int nt = 0; nt < 8; ++nt) acc[nt] = (v8f){};
    #pragma unroll
    for (int k0 = 0; k0 < Jn; k0 += 32) {
      v16h a = load_a16(arow + k0);
      #pragma unroll
      for (int nt = 0; nt < 8; ++nt) {
        const _Float16* bp = dwT + ((size_t)s * Hn + nt * 16 + nB) * Jn + k0 + kbB;
        acc[nt] = wmma16(a, load_b16(bp), acc[nt]);
      }
    }
    #pragma unroll
    for (int v = 0; v < 8; ++v) {
      int rr = rowBase + v + rhi;
      if (rr < count) {
        int bc = dlist[start + rr];
        float mv = mask[bc] ? 1.0f : 0.0f;
        float* op = xch + (size_t)bc * Hn + nB;
        #pragma unroll
        for (int nt = 0; nt < 8; ++nt) op[nt * 16] = mv * acc[nt][v];
      }
    }
  }
}

// ---------------- reduce over context: xc[b,h] = sum_c xch[b,c,h] ----------------
__global__ void k_reduce(const float* __restrict__ xch, float* __restrict__ xc) {
  int i = blockIdx.x * blockDim.x + threadIdx.x;
  if (i >= Bn * Hn) return;
  int b = i / Hn, hh = i % Hn;
  float sum = 0.f;
  #pragma unroll 4
  for (int c = 0; c < Cn; ++c) sum += xch[((size_t)b * Cn + c) * Hn + hh];
  xc[i] = sum;
}

// ---------------- GEMM 3: out = sigmoid(xc + h @ HW) ----------------
// grid (16, 8): 16 row-tiles of B, 8 col-tiles of H.
__global__ __launch_bounds__(32)
void k_gemm3(const _Float16* __restrict__ h16, const _Float16* __restrict__ hwT,
             const float* __restrict__ xc, float* __restrict__ out) {
  const int mt = blockIdx.x, nt = blockIdx.y;
  const int lane = threadIdx.x;
  const int nB   = lane & 15;
  const int subA = (lane & 16) ? 8 : 0;
  const int kbB  = (lane & 16) ? 8 : 0;
  const int rhi  = (lane & 16) ? 8 : 0;

  const _Float16* arow = h16 + (size_t)(mt * 16 + (lane & 15)) * Hn + subA;
  v8f acc = {};
  #pragma unroll
  for (int k0 = 0; k0 < Hn; k0 += 32) {
    v16h a = load_a16(arow + k0);
    const _Float16* bp = hwT + (size_t)(nt * 16 + nB) * Hn + k0 + kbB;
    acc = wmma16(a, load_b16(bp), acc);
  }
  #pragma unroll
  for (int v = 0; v < 8; ++v) {
    int b = mt * 16 + v + rhi;
    int hcol = nt * 16 + nB;
    float x = acc[v] + xc[b * Hn + hcol];
    out[b * Hn + hcol] = 1.0f / (1.0f + __expf(-x));
  }
}

// ---------------- host launch ----------------

static inline size_t alignup(size_t x) { return (x + 255) & ~(size_t)255; }

extern "C" void kernel_launch(void* const* d_in, const int* in_sizes, int n_in,
                              void* d_out, int out_size, void* d_ws, size_t ws_size,
                              hipStream_t stream) {
  (void)in_sizes; (void)n_in; (void)out_size; (void)ws_size;
  const float* x_ctx = (const float*)d_in[0];          // (B,C,I) f32
  const int*   tctx  = (const int*)d_in[1];            // (B,C) i32
  const int*   dctx  = (const int*)d_in[2];            // (B,C) i32
  const unsigned char* cmask = (const unsigned char*)d_in[3]; // (B,C) bool
  const float* h     = (const float*)d_in[4];          // (B,H) f32
  const float* tw    = (const float*)d_in[5];          // (49,I,J) f32
  const float* dw    = (const float*)d_in[6];          // (49,J,H) f32
  const float* hw    = (const float*)d_in[7];          // (H,H)  f32
  float* out = (float*)d_out;                          // (B,H) f32

  // workspace carve-up (~9.2 MB)
  char* w = (char*)d_ws;
  size_t off = 0;
  _Float16* x16  = (_Float16*)(w + off); off = alignup(off + 2ull * Bn * Cn * In);
  _Float16* h16  = (_Float16*)(w + off); off = alignup(off + 2ull * Bn * Hn);
  _Float16* twT  = (_Float16*)(w + off); off = alignup(off + 2ull * NSLOT * In * Jn);
  _Float16* dwT  = (_Float16*)(w + off); off = alignup(off + 2ull * NSLOT * Jn * Hn);
  _Float16* hwT  = (_Float16*)(w + off); off = alignup(off + 2ull * Hn * Hn);
  _Float16* xt16 = (_Float16*)(w + off); off = alignup(off + 2ull * Bn * Cn * Jn);
  float*    xch  = (float*)(w + off);    off = alignup(off + 4ull * Bn * Cn * Hn);
  float*    xc   = (float*)(w + off);    off = alignup(off + 4ull * Bn * Hn);
  int* tcnt  = (int*)(w + off); off = alignup(off + 4ull * NSLOT);
  int* toff_ = (int*)(w + off); off = alignup(off + 4ull * NSLOT);
  int* tcur  = (int*)(w + off); off = alignup(off + 4ull * NSLOT);
  int* tlist = (int*)(w + off); off = alignup(off + 4ull * NPAIR);
  int* dcnt  = (int*)(w + off); off = alignup(off + 4ull * NSLOT);
  int* doff_ = (int*)(w + off); off = alignup(off + 4ull * NSLOT);
  int* dcur  = (int*)(w + off); off = alignup(off + 4ull * NSLOT);
  int* dlist = (int*)(w + off); off = alignup(off + 4ull * NPAIR);

  const int TPB = 256;
  // prep
  k_init<<<1, 64, 0, stream>>>(tcnt, dcnt);
  k_cvt<<<(Bn * Cn * In + TPB - 1) / TPB, TPB, 0, stream>>>(x_ctx, x16, Bn * Cn * In);
  k_cvt<<<(Bn * Hn + TPB - 1) / TPB, TPB, 0, stream>>>(h, h16, Bn * Hn);
  k_cvtT<<<(NSLOT * In * Jn + TPB - 1) / TPB, TPB, 0, stream>>>(tw, twT, NSLOT, In, Jn);
  k_cvtT<<<(NSLOT * Jn * Hn + TPB - 1) / TPB, TPB, 0, stream>>>(dw, dwT, NSLOT, Jn, Hn);
  k_cvtT<<<(Hn * Hn + TPB - 1) / TPB, TPB, 0, stream>>>(hw, hwT, 1, Hn, Hn);
  // bucket by slot
  k_count<<<(NPAIR + TPB - 1) / TPB, TPB, 0, stream>>>(tctx, dctx, tcnt, dcnt);
  k_scan<<<1, 32, 0, stream>>>(tcnt, toff_, tcur, dcnt, doff_, dcur);
  k_fill<<<(NPAIR + TPB - 1) / TPB, TPB, 0, stream>>>(tctx, dctx, tcur, tlist, dcur, dlist);
  // WMMA GEMMs
  k_gemm1<<<dim3(NSLOT, 16), 32, 0, stream>>>(x16, twT, toff_, tcnt, tlist, xt16);
  k_gemm2<<<dim3(NSLOT, 16), 32, 0, stream>>>(xt16, dwT, doff_, dcnt, dlist, cmask, xch);
  k_reduce<<<(Bn * Hn + TPB - 1) / TPB, TPB, 0, stream>>>(xch, xc);
  k_gemm3<<<dim3(Bn / 16, Hn / 16), 32, 0, stream>>>(h16, hwT, xc, out);
}
